// GRU_56745107915060
// MI455X (gfx1250) — matrix operand order repaired and checked by reading, not compiled, not run on hardware
//
#include <hip/hip_runtime.h>
#include <hip/hip_bf16.h>
#include <stdint.h>

// GRU recurrence, MI455X (gfx1250, wave32, WMMA bf16 16x16x32).
// 8 persistent workgroups x 256 threads; weights LDS-resident; global
// atomic split-barrier (2/step); h carried in registers across steps.
// Per-step h / r*h global->LDS staging uses GLOBAL_LOAD_ASYNC_TO_LDS_B128
// (ASYNCcnt); phase-2 weight fragments are hoisted into VGPRs for all
// 2048 steps (loop-invariant B operands).

#define SEQ     2048
#define BATCH   64
#define INDIM   128
#define HID     256
#define KTOT    (INDIM + HID)     // 384
#define NWG     8
#define SLICE   (HID / NWG)       // 32 output channels per WG
#define THREADS 256

typedef __bf16 bf16_t;
typedef __attribute__((ext_vector_type(16))) __bf16 v16bf;
typedef __attribute__((ext_vector_type(8)))  __bf16 v8bf;
typedef __attribute__((ext_vector_type(4)))  __bf16 v4bf;
typedef __attribute__((ext_vector_type(8)))  float  v8f;
typedef __attribute__((ext_vector_type(4)))  float  v4f;

// workspace layout (bytes)
#define WB_BYTES   (3 * HID * KTOT * 2)   // 589824: permuted bf16 weights [3][256][384]
#define HB_BYTES   (BATCH * HID * 2)      // 32768 : h_{t-1} bf16 [64][256]
#define RH_BYTES   (BATCH * HID * 2)      // 32768 : r*h bf16 [64][256]
#define CTR_BYTES  64
// dynamic LDS: A [64][384] bf16 + weight slice [3][32][384] bf16
#define SMEM_BYTES (BATCH * KTOT * 2 + 3 * SLICE * KTOT * 2)  // 122880

// ---------------------------------------------------------------- prep kernel
// Permute Wz/Wr columns so every gate reads A = [h(256) | x(128)]:
//   gates z,r: out col k -> orig col (k<HID ? INDIM+k : k-HID)
//   gate  i  : identity (Wi already is [h | x])
// Also seed the bf16 h buffer from h0.
__global__ void gru_prep_kernel(const float* __restrict__ Wz,
                                const float* __restrict__ Wr,
                                const float* __restrict__ Wi,
                                const float* __restrict__ h0,
                                bf16_t* __restrict__ Wb,
                                bf16_t* __restrict__ Hb) {
    int idx = blockIdx.x * blockDim.x + threadIdx.x;
    const int total = 3 * HID * KTOT;
    if (idx < total) {
        int g   = idx / (HID * KTOT);
        int rem = idx % (HID * KTOT);
        int n   = rem / KTOT;
        int k   = rem % KTOT;
        const float* W = (g == 0) ? Wz : (g == 1) ? Wr : Wi;
        int ksrc = (g == 2) ? k : (k < HID ? (INDIM + k) : (k - HID));
        Wb[idx] = (bf16_t)W[n * KTOT + ksrc];
    }
    if (idx < BATCH * HID) Hb[idx] = (bf16_t)h0[idx];
}

// ---------------------------------------------------------------- helpers
__device__ __forceinline__ v16bf load_afrag(const bf16_t* sA, int m0, int lr,
                                            int half, int k0) {
    // 16-bit A 16x32 (ISA 7.12.2): lanes 0-15 K={k0..k0+7, k0+16..k0+23},
    // lanes 16-31 K shifted by +8.
    const bf16_t* p = sA + (m0 + lr) * KTOT + k0 + half * 8;
    v8bf lo = *(const v8bf*)p;
    v8bf hi = *(const v8bf*)(p + 16);
    return __builtin_shufflevector(lo, hi, 0,1,2,3,4,5,6,7,8,9,10,11,12,13,14,15);
}

__device__ __forceinline__ v16bf load_bfrag(const bf16_t* sW, int g, int n0,
                                            int lr, int half, int k0) {
    // B 32x16: lanes 0-15 hold K={k0..k0+7, k0+16..k0+23} of column n0+lr (== W row),
    // lanes 16-31 K shifted by +8.
    const bf16_t* p = sW + (g * SLICE + n0 + lr) * KTOT + k0 + half * 8;
    v8bf lo = *(const v8bf*)p;
    v8bf hi = *(const v8bf*)(p + 16);
    return __builtin_shufflevector(lo, hi, 0,1,2,3,4,5,6,7,8,9,10,11,12,13,14,15);
}

// CDNA5 async DMA: global -> LDS, no VGPR staging, tracked by ASYNCcnt.
// lds_off is a byte offset into the workgroup's LDS allocation (our single
// extern __shared__ block starts at offset 0).
__device__ __forceinline__ void async_ld_lds_b128(uint32_t lds_off, const void* g) {
    asm volatile("global_load_async_to_lds_b128 %0, %1, off"
                 :: "v"(lds_off), "v"(g)
                 : "memory");
}
__device__ __forceinline__ void wait_asynccnt0() {
    asm volatile("s_wait_asynccnt 0x0" ::: "memory");
}

__device__ __forceinline__ void global_barrier(unsigned* ctr, unsigned& gen) {
    __syncthreads();
    __threadfence();
    if (threadIdx.x == 0) {
        atomicAdd(ctr, 1u);
        const unsigned target = (gen + 1u) * (unsigned)NWG;
        while (__hip_atomic_load(ctr, __ATOMIC_ACQUIRE,
                                 __HIP_MEMORY_SCOPE_AGENT) < target) {
            __builtin_amdgcn_s_sleep(1);
        }
        __threadfence();
    }
    ++gen;
    __syncthreads();
}

__device__ __forceinline__ float sigmoid_f(float x) {
    return 1.0f / (1.0f + __expf(-x));
}

// ---------------------------------------------------------------- main kernel
__global__ void __launch_bounds__(THREADS, 1)
gru_recurrence_kernel(const float* __restrict__ x,
                      const float* __restrict__ h0,
                      const bf16_t* __restrict__ Wb,
                      bf16_t* __restrict__ Hb,
                      bf16_t* __restrict__ RH,
                      unsigned* __restrict__ ctr,
                      float* __restrict__ out) {
    extern __shared__ __align__(16) char smem[];
    bf16_t* sA = (bf16_t*)smem;                              // [64][384] @ LDS 0
    bf16_t* sW = (bf16_t*)(smem + BATCH * KTOT * 2);         // [3][32][384]

    const int tid   = threadIdx.x;
    const int wg    = blockIdx.x;
    const int wave  = tid >> 5;
    const int lane  = tid & 31;
    const int half  = lane >> 4;
    const int lr    = lane & 15;

    // one 16x16 output tile per wave: 4 M-tiles x 2 N-tiles
    const int m0    = (wave >> 1) * 16;
    const int n0    = (wave & 1) * 16;     // within slice
    const int nbase = wg * SLICE;          // global channel base
    const int ccol  = nbase + n0 + lr;     // this lane's output channel

    // ---- one-time: weight slice -> LDS (72 KB, stays resident) ----
    {
        const int chunks = 3 * SLICE * KTOT / 8;   // 16B chunks
        const int per_row = KTOT / 8;              // 48
        for (int i = tid; i < chunks; i += THREADS) {
            int row = i / per_row;                 // 0..95 => [g][n]
            int col = i % per_row;
            int g = row / SLICE, n = row % SLICE;
            *(v8bf*)(sW + row * KTOT + col * 8) =
                *(const v8bf*)(Wb + ((size_t)(g * HID + nbase + n) * KTOT) + col * 8);
        }
    }

    // ---- carry h in registers: this wave's fragment of h ----
    float hreg[8];
#pragma unroll
    for (int v = 0; v < 8; ++v)
        hreg[v] = h0[(m0 + v + 8 * half) * HID + ccol];

    __syncthreads();

    // ---- hoist phase-2 (candidate-gate) weight fragments into VGPRs:
    // loop-invariant across all 2048 timesteps (12 x v16bf = 96 VGPRs).
    v16bf bi[KTOT / 32];
#pragma unroll
    for (int kk = 0; kk < KTOT / 32; ++kk)
        bi[kk] = load_bfrag(sW, 2, n0, lr, half, kk * 32);

    unsigned gen = 0;
    for (int t = 0; t < SEQ; ++t) {
        const float* xt = x + (size_t)t * BATCH * INDIM;

        // prefetch next timestep's x while we compute this one
        if (t + 1 < SEQ) {
            const char* nx = (const char*)(xt + BATCH * INDIM);
            __builtin_prefetch(nx + tid * 128, 0, 0);
        }

        // ---- build A = [ h_{t-1} bf16 | x_t -> bf16 ] in LDS ----
        // h part: async DMA global -> LDS (8 x b128 per thread, ASYNCcnt)
#pragma unroll
        for (int c = 0; c < 8; ++c) {
            int i = tid + c * THREADS;
            int row = i >> 5, col = i & 31;                      // 32 chunks/row
            async_ld_lds_b128((uint32_t)(row * (KTOT * 2) + col * 16),
                              (const char*)Hb + row * (HID * 2) + col * 16);
        }
        // x part: fp32 -> bf16 conversion on VALU (overlaps async DMA)
        for (int i = tid; i < BATCH * INDIM / 4; i += THREADS) { // 32 f4/row
            int row = i >> 5, col = i & 31;
            v4f xv = *(const v4f*)(xt + row * INDIM + col * 4);
            *(v4bf*)(sA + row * KTOT + HID + col * 4) =
                __builtin_convertvector(xv, v4bf);
        }
        wait_asynccnt0();
        __syncthreads();

        // ---- phase 1: z and r GEMMs (shared A fragment) ----
        v8f accz = {}, accr = {};
        for (int k0 = 0; k0 < KTOT; k0 += 32) {
            v16bf af  = load_afrag(sA, m0, lr, half, k0);
            v16bf bzf = load_bfrag(sW, 0, n0, lr, half, k0);
            v16bf brf = load_bfrag(sW, 1, n0, lr, half, k0);
            accz = __builtin_amdgcn_wmma_f32_16x16x32_bf16(
                false, af, false, bzf, (short)0, accz, false, false);
            accr = __builtin_amdgcn_wmma_f32_16x16x32_bf16(
                false, af, false, brf, (short)0, accr, false, false);
        }

        float zs[8];
#pragma unroll
        for (int v = 0; v < 8; ++v) {
            int m = m0 + v + 8 * half;
            zs[v] = sigmoid_f(accz[v]);
            float rv = sigmoid_f(accr[v]);
            RH[m * HID + ccol] = (bf16_t)(rv * hreg[v]);   // publish r*h slice
        }

        global_barrier(ctr, gen);   // all r*h slices visible

        // ---- overwrite A[:,0:256] with r*h (async DMA again) ----
#pragma unroll
        for (int c = 0; c < 8; ++c) {
            int i = tid + c * THREADS;
            int row = i >> 5, col = i & 31;
            async_ld_lds_b128((uint32_t)(row * (KTOT * 2) + col * 16),
                              (const char*)RH + row * (HID * 2) + col * 16);
        }
        wait_asynccnt0();
        __syncthreads();

        // ---- phase 2: candidate GEMM (B operands already in VGPRs) ----
        v8f acci = {};
        for (int kk = 0; kk < KTOT / 32; ++kk) {
            v16bf af = load_afrag(sA, m0, lr, half, kk * 32);
            acci = __builtin_amdgcn_wmma_f32_16x16x32_bf16(
                false, af, false, bi[kk], (short)0, acci, false, false);
        }

        float* outp = out + (size_t)t * BATCH * HID;
#pragma unroll
        for (int v = 0; v < 8; ++v) {
            int m = m0 + v + 8 * half;
            float it = tanhf(acci[v]);
            float hn = (1.0f - zs[v]) * hreg[v] + zs[v] * it;
            hreg[v] = hn;
            outp[m * HID + ccol] = hn;               // fp32 result (= h_full[t])
            Hb[m * HID + ccol]   = (bf16_t)hn;       // bf16 for next step's A
        }

        global_barrier(ctr, gen);   // h_t fully published before next step
    }
}

// ---------------------------------------------------------------- launch
extern "C" void kernel_launch(void* const* d_in, const int* in_sizes, int n_in,
                              void* d_out, int out_size, void* d_ws, size_t ws_size,
                              hipStream_t stream) {
    const float* x  = (const float*)d_in[0];   // (2048, 64, 128)
    const float* h0 = (const float*)d_in[1];   // (64, 256)
    const float* Wz = (const float*)d_in[2];   // (256, 384)
    const float* Wr = (const float*)d_in[3];
    const float* Wi = (const float*)d_in[4];
    float* out = (float*)d_out;                // (2048, 64, 256)

    uintptr_t base = (uintptr_t)d_ws;
    bf16_t*   Wb  = (bf16_t*)base;
    bf16_t*   Hb  = (bf16_t*)(base + WB_BYTES);
    bf16_t*   RH  = (bf16_t*)(base + WB_BYTES + HB_BYTES);
    unsigned* ctr = (unsigned*)(base + WB_BYTES + HB_BYTES + RH_BYTES);

    hipMemsetAsync(ctr, 0, CTR_BYTES, stream);   // reset barrier counter

    int prep_threads = 3 * HID * KTOT;           // 294912
    gru_prep_kernel<<<(prep_threads + THREADS - 1) / THREADS, THREADS, 0, stream>>>(
        Wz, Wr, Wi, h0, Wb, Hb);

    gru_recurrence_kernel<<<NWG, THREADS, SMEM_BYTES, stream>>>(
        x, h0, Wb, Hb, RH, ctr, out);
}
